// DualAttention_40630390620992
// MI455X (gfx1250) — matrix operand order repaired and checked by reading, not compiled, not run on hardware
//
#include <hip/hip_runtime.h>
#include <math.h>

typedef __attribute__((ext_vector_type(16))) __bf16 v16bf;
typedef __attribute__((ext_vector_type(8)))  float  v8f;

#define B_ 64
#define S_ 2048
#define H_ 512
#define F_ 128

__device__ __forceinline__ unsigned short f32_to_bf16_bits(float f) {
    unsigned int x = __float_as_uint(f);
    unsigned int r = x + 0x7FFFu + ((x >> 16) & 1u);   // round-to-nearest-even
    return (unsigned short)(r >> 16);
}

// branch-free tanh: 1 - 2/(e^{2x}+1); clamp keeps exp finite
__device__ __forceinline__ float fast_tanh(float x) {
    float xc = fminf(fmaxf(x, -15.0f), 15.0f);
    float e = __expf(2.0f * xc);
    return 1.0f - 2.0f / (e + 1.0f);
}

union FragBF { v16bf v; uint4 q[2]; unsigned short u[16]; };

// ---------------- prep: weights f32 -> bf16, TRANSPOSED (Wt[n][k]) ----------------
__global__ __launch_bounds__(256)
void k_convert_w(const float* __restrict__ W_enc, const float* __restrict__ W_z,
                 unsigned short* __restrict__ WencT, unsigned short* __restrict__ WzT) {
    int idx = blockIdx.x * 256 + threadIdx.x;
    if (idx < H_ * H_) {                      // W_enc[k][n] -> WencT[n][k]
        int k = idx >> 9, n = idx & 511;
        WencT[n * H_ + k] = f32_to_bf16_bits(W_enc[idx]);
    }
    int j = idx - H_ * H_;
    if (j >= 0 && j < F_ * H_) {              // W_z[k][n] -> WzT[n][k]
        int k = j >> 9, n = j & 511;
        WzT[n * F_ + k] = f32_to_bf16_bits(W_z[j]);
    }
}

// ---------------- prep: proj_in_a / proj_in_b (tiny GEMMs) ----------------
__global__ __launch_bounds__(256)
void k_proj_in(const float* __restrict__ input,
               const float* __restrict__ W_a, const float* __restrict__ b_a,
               const float* __restrict__ W_b, const float* __restrict__ b_b,
               float* __restrict__ pa, float* __restrict__ pb) {
    int idx = blockIdx.x * 256 + threadIdx.x;       // 0..65535
    int which = idx >> 15;                          // 0 -> a, 1 -> b
    int lid = idx & 32767;
    int b = lid >> 9, h = lid & 511;
    const float* W    = which ? W_b : W_a;
    const float* bias = which ? b_b : b_a;
    const float* in = input + b * H_;
    float acc = bias[h];
    for (int k = 0; k < H_; ++k) acc += in[k] * W[k * H_ + h];
    (which ? pb : pa)[lid] = fast_tanh(acc);
}

// ---------------- fused WMMA scoring kernel (32 rows / block) ----------------
__global__ __launch_bounds__(256)
void k_scores(const float* __restrict__ context,
              const float* __restrict__ input_z,
              const unsigned short* __restrict__ WencT,   // [512][512] bf16, N-major
              const unsigned short* __restrict__ WzT,     // [512][128] bf16, N-major
              const float* __restrict__ b_enc, const float* __restrict__ b_z,
              const float* __restrict__ pa,    const float* __restrict__ pb,
              float* __restrict__ scores) {
    // padded to 40 halves/row: 80B stride = 16B-aligned b128 reads, 20-dword bank stride
    __shared__ __align__(16) unsigned short Alds[32][40];   // 32 rows x 32 K (bf16)
    __shared__ __align__(16) unsigned short Wlds[512][40];  // 512 N x 32 K (bf16)
    __shared__ float sh_part[8][32];

    const int tid  = threadIdx.x;
    const int wave = tid >> 5;
    const int lane = tid & 31;
    const int half = lane >> 4;
    const int m    = lane & 15;
    const int row0 = blockIdx.x * 32;            // 32-row M-tile, all same b
    const int b    = row0 / S_;
    const int colbase = wave * 64;               // this wave owns 4 N-tiles

    unsigned int* WldsU = (unsigned int*)&Wlds[0][0];

    float sacc[2][8];
    for (int st = 0; st < 2; ++st)
        for (int v = 0; v < 8; ++v) sacc[st][v] = 0.0f;

    auto run_phase = [&](const float* Asrc, int ldA, const unsigned short* Wt,
                         int Kdim, v8f acc[2][4]) {
        const unsigned int* Wt32 = (const unsigned int*)Wt;
        const int kd2 = Kdim >> 1;               // dwords per Wt row
        const int ksteps = Kdim / 32;
        // per-thread A-tile coords: one float4 load covers the whole 32x32 tile
        const int ar = tid >> 3;                 // 0..31
        const int akq = tid & 7;                 // quad within row (k = akq*4..+3)
        for (int ks = 0; ks < ksteps; ++ks) {
            const int k0 = ks * 32;
            __syncthreads();
            // stage A: 32x32 f32 -> bf16, one float4 + one packed b64 store per thread
            {
                float4 f = *(const float4*)&Asrc[(row0 + ar) * ldA + k0 + akq * 4];
                uint2 p;
                p.x = (unsigned)f32_to_bf16_bits(f.x) |
                      ((unsigned)f32_to_bf16_bits(f.y) << 16);
                p.y = (unsigned)f32_to_bf16_bits(f.z) |
                      ((unsigned)f32_to_bf16_bits(f.w) << 16);
                *(uint2*)&Alds[ar][akq * 4] = p;
            }
            // stage Wt chunk: 512 rows x 16 dwords, as 8 x b128 per thread
            for (int i = 0; i < 8; ++i) {
                int idx = tid + (i << 8);        // 0..2047
                int n = idx >> 2, q = idx & 3;
                uint4 val = *(const uint4*)(Wt32 + n * kd2 + (k0 >> 1) + q * 4);
                *(uint4*)(WldsU + n * 20 + q * 4) = val;
            }
            // prefetch next K chunk of Wt into caches
            if (ks + 1 < ksteps)
                __builtin_prefetch(Wt32 + ((unsigned)tid << 5) + ((k0 + 32) >> 1), 0, 1);
            __syncthreads();
            // B fragments: 2 contiguous b128 reads each, reused across both strips
            FragBF bf[4];
            for (int t = 0; t < 4; ++t) {
                int n = colbase + t * 16 + m;
                bf[t].q[0] = *(const uint4*)&Wlds[n][half * 8];
                bf[t].q[1] = *(const uint4*)&Wlds[n][16 + half * 8];
            }
            for (int st = 0; st < 2; ++st) {
                FragBF af;
                int r = st * 16 + m;
                af.q[0] = *(const uint4*)&Alds[r][half * 8];
                af.q[1] = *(const uint4*)&Alds[r][16 + half * 8];
                for (int t = 0; t < 4; ++t)
                    acc[st][t] = __builtin_amdgcn_wmma_f32_16x16x32_bf16(
                        false, af.v, false, bf[t].v, (short)0, acc[st][t], false, false);
            }
        }
    };

    auto epilogue = [&](v8f acc[2][4], const float* bias, const float* pin) {
        for (int t = 0; t < 4; ++t) {
            int n = colbase + t * 16 + m;
            float bv = bias[n];
            float pv = pin[b * H_ + n];
            for (int st = 0; st < 2; ++st)
                for (int v = 0; v < 8; ++v)
                    sacc[st][v] += fast_tanh(acc[st][t][v] + bv) * pv;
        }
    };

    {   // encoder path: K = 512
        v8f acc[2][4];
        for (int st = 0; st < 2; ++st)
            for (int t = 0; t < 4; ++t)
                for (int v = 0; v < 8; ++v) acc[st][t][v] = 0.0f;
        run_phase(context, H_, WencT, H_, acc);
        epilogue(acc, b_enc, pa);
    }
    {   // field-rep path: K = 128
        v8f acc[2][4];
        for (int st = 0; st < 2; ++st)
            for (int t = 0; t < 4; ++t)
                for (int v = 0; v < 8; ++v) acc[st][t][v] = 0.0f;
        run_phase(input_z, F_, WzT, F_, acc);
        epilogue(acc, b_z, pb);
    }

    // reduce over the 16 lanes of each half (wave32; width=16 groups)
    for (int st = 0; st < 2; ++st)
        for (int v = 0; v < 8; ++v) {
            float s = sacc[st][v];
            s += __shfl_xor(s, 8, 16);
            s += __shfl_xor(s, 4, 16);
            s += __shfl_xor(s, 2, 16);
            s += __shfl_xor(s, 1, 16);
            sacc[st][v] = s;
        }
    if (m == 0)
        for (int st = 0; st < 2; ++st)
            for (int v = 0; v < 8; ++v)
                sh_part[wave][st * 16 + half * 8 + v] = sacc[st][v];
    __syncthreads();
    if (tid < 32) {                              // deterministic cross-wave sum
        float s = 0.0f;
        for (int w = 0; w < 8; ++w) s += sh_part[w][tid];
        scores[row0 + tid] = s;
    }
}

// ---------------- softmax(scores_a + scores_b) -> gamma ----------------
__global__ __launch_bounds__(256)
void k_softmax(const float* __restrict__ scores, float* __restrict__ gamma) {
    __shared__ float red[256];
    int b = blockIdx.x, tid = threadIdx.x;
    const float* s = scores + b * S_;
    float loc[8], mx = -1e30f;
    for (int i = 0; i < 8; ++i) { loc[i] = s[tid + i * 256]; mx = fmaxf(mx, loc[i]); }
    red[tid] = mx; __syncthreads();
    for (int off = 128; off > 0; off >>= 1) {
        if (tid < off) red[tid] = fmaxf(red[tid], red[tid + off]);
        __syncthreads();
    }
    mx = red[0]; __syncthreads();
    float sum = 0.0f;
    for (int i = 0; i < 8; ++i) { loc[i] = __expf(loc[i] - mx); sum += loc[i]; }
    red[tid] = sum; __syncthreads();
    for (int off = 128; off > 0; off >>= 1) {
        if (tid < off) red[tid] += red[tid + off];
        __syncthreads();
    }
    float inv = 1.0f / red[0];
    float* g = gamma + b * S_;
    for (int i = 0; i < 8; ++i) g[tid + i * 256] = loc[i] * inv;
}

// ---------------- weighted context: wc[b,h] = sum_s gamma[b,s]*ctx[b,s,h] ----------------
__global__ __launch_bounds__(256)
void k_weighted(const float* __restrict__ context, const float* __restrict__ gamma,
                float* __restrict__ wc) {
    int b = blockIdx.x, tid = threadIdx.x;
    const float2* ctx2 = (const float2*)(context + (size_t)b * S_ * H_);
    const float* g = gamma + b * S_;
    float a0 = 0.0f, a1 = 0.0f;
    for (int s = 0; s < S_; ++s) {
        float gv = g[s];
        float2 c = ctx2[s * 256 + tid];
        a0 += gv * c.x;
        a1 += gv * c.y;
    }
    float2* out = (float2*)(wc + b * H_);
    out[tid] = make_float2(a0, a1);
}

// ---------------- h_tilde = tanh([wc, input] @ W_out) ----------------
__global__ __launch_bounds__(256)
void k_out(const float* __restrict__ wc, const float* __restrict__ input,
           const float* __restrict__ W_out, float* __restrict__ h_tilde) {
    int idx = blockIdx.x * 256 + threadIdx.x;    // 0..32767
    int b = idx >> 9, h = idx & 511;
    const float* w0  = W_out + h;
    const float* wcb = wc + b * H_;
    const float* inb = input + b * H_;
    float acc = 0.0f;
    for (int k = 0; k < H_; ++k) acc += wcb[k] * w0[k * H_];
    for (int k = 0; k < H_; ++k) acc += inb[k] * w0[(H_ + k) * H_];
    h_tilde[idx] = fast_tanh(acc);
}

extern "C" void kernel_launch(void* const* d_in, const int* in_sizes, int n_in,
                              void* d_out, int out_size, void* d_ws, size_t ws_size,
                              hipStream_t stream) {
    (void)in_sizes; (void)n_in; (void)out_size; (void)ws_size;
    const float* input   = (const float*)d_in[0];
    const float* context = (const float*)d_in[1];
    const float* input_z = (const float*)d_in[2];
    const float* W_enc   = (const float*)d_in[3];
    const float* b_enc   = (const float*)d_in[4];
    const float* W_a     = (const float*)d_in[5];
    const float* b_a     = (const float*)d_in[6];
    const float* W_b     = (const float*)d_in[7];
    const float* b_b     = (const float*)d_in[8];
    const float* W_z     = (const float*)d_in[9];
    const float* b_z     = (const float*)d_in[10];
    const float* W_out   = (const float*)d_in[11];

    float* h_tilde = (float*)d_out;               // [B,H]
    float* gamma   = (float*)d_out + B_ * H_;     // [B,S]

    char* ws = (char*)d_ws;                       // ~1.5 MB total
    unsigned short* WencT = (unsigned short*)(ws);              // 512 KB  [512][512]
    unsigned short* WzT   = (unsigned short*)(ws + 524288);     // 128 KB  [512][128]
    float* pa     = (float*)(ws + 655360);                      // 128 KB
    float* pb     = (float*)(ws + 786432);                      // 128 KB
    float* scores = (float*)(ws + 917504);                      // 512 KB
    float* wc     = (float*)(ws + 1441792);                     // 128 KB

    k_convert_w<<<1280, 256, 0, stream>>>(W_enc, W_z, WencT, WzT);
    k_proj_in <<<256, 256, 0, stream>>>(input, W_a, b_a, W_b, b_b, pa, pb);
    k_scores  <<<(B_ * S_) / 32, 256, 0, stream>>>(context, input_z, WencT, WzT,
                                                   b_enc, b_z, pa, pb, scores);
    k_softmax <<<B_, 256, 0, stream>>>(scores, gamma);
    k_weighted<<<B_, 256, 0, stream>>>(context, gamma, wc);
    k_out     <<<B_ * H_ / 256, 256, 0, stream>>>(wc, input, W_out, h_tilde);
}
